// Block_1915555414235
// MI455X (gfx1250) — matrix operand order repaired and checked by reading, not compiled, not run on hardware
//
#include <hip/hip_runtime.h>

// ---------------------------------------------------------------------------
// Transformer block (pre-norm causal MHA + FFN) for gfx1250 (MI455X).
// All GEMMs + attention run on v_wmma_f32_16x16x32_bf16 (f32 accumulate).
// ---------------------------------------------------------------------------

#define BB   128
#define TT   256
#define CC   384
#define HH   6
#define HSZ  64
#define DFFN 1536
#define MM   (BB * TT)   // 32768 rows

typedef __bf16 bf16x16 __attribute__((ext_vector_type(16)));
typedef float  f32x8   __attribute__((ext_vector_type(8)));

union Frag {
  bf16x16        v;
  unsigned int   u[8];
  unsigned short s[16];
};

// K index held by packed pair r (0..7) of an A-fragment, for lane-half `half`.
// (ISA 7.12.2, 16-bit A 16x32: V0..3 -> K 0..7 / 8..15, V4..7 -> K 16..23 / 24..31)
__device__ __forceinline__ int kidx(int r, int half) {
  return ((r < 4) ? (2 * r) : (8 + 2 * r)) + 8 * half;
}

__device__ __forceinline__ unsigned short f2bf(float f) {
  unsigned int u = __float_as_uint(f);
  u += 0x7fffu + ((u >> 16) & 1u);   // round-to-nearest-even
  return (unsigned short)(u >> 16);
}

__device__ __forceinline__ f32x8 zero8() {
  f32x8 z;
#pragma unroll
  for (int i = 0; i < 8; ++i) z[i] = 0.0f;
  return z;
}

__device__ __forceinline__ f32x8 wmma_bf16(const Frag& a, const Frag& b, f32x8 c) {
  return __builtin_amdgcn_wmma_f32_16x16x32_bf16(
      /*neg_a=*/false, a.v, /*neg_b=*/false, b.v,
      /*c_mod=*/(short)0, c, /*reuse_a=*/false, /*reuse_b=*/false);
}

// ---------------------------------------------------------------------------
// f32 -> bf16 weight repack into row-major (K, N).
// out[k*N + n] = in[k*sk + (n>>6)*sh + (n&63)*sd]
// Handles plain (K,N) weights and the (H,C,HS) QKV weights in one kernel.
// ---------------------------------------------------------------------------
__global__ void convert_weight(const float* __restrict__ w,
                               unsigned short* __restrict__ o,
                               int K, int N, int sk, int sh, int sd) {
  int i = blockIdx.x * blockDim.x + threadIdx.x;
  if (i >= K * N) return;
  int k = i / N, n = i % N;
  o[i] = f2bf(w[(size_t)k * sk + (size_t)(n >> 6) * sh + (size_t)(n & 63) * sd]);
}

// ---------------------------------------------------------------------------
// LayerNorm over C=384, one row per 128-thread block, bf16 output.
// ---------------------------------------------------------------------------
__global__ void __launch_bounds__(128) layernorm_bf16(
    const float* __restrict__ x,
    const float* __restrict__ g,
    const float* __restrict__ be,
    unsigned short* __restrict__ o) {
  __shared__ float red[128];
  const int row = blockIdx.x;
  const int t   = threadIdx.x;
  const float* xr = x + (size_t)row * CC;
  float v0 = xr[t], v1 = xr[t + 128], v2 = xr[t + 256];

  red[t] = v0 + v1 + v2;
  __syncthreads();
  for (int off = 64; off > 0; off >>= 1) {
    if (t < off) red[t] += red[t + off];
    __syncthreads();
  }
  float mu = red[0] * (1.0f / CC);
  __syncthreads();

  float d0 = v0 - mu, d1 = v1 - mu, d2 = v2 - mu;
  red[t] = d0 * d0 + d1 * d1 + d2 * d2;
  __syncthreads();
  for (int off = 64; off > 0; off >>= 1) {
    if (t < off) red[t] += red[t + off];
    __syncthreads();
  }
  float rstd = rsqrtf(red[0] * (1.0f / CC) + 1e-5f);

  unsigned short* orow = o + (size_t)row * CC;
  orow[t]       = f2bf(d0 * rstd * g[t]       + be[t]);
  orow[t + 128] = f2bf(d1 * rstd * g[t + 128] + be[t + 128]);
  orow[t + 256] = f2bf(d2 * rstd * g[t + 256] + be[t + 256]);
}

// ---------------------------------------------------------------------------
// WMMA GEMM: out = act(oscale * (A(MxK,bf16) * W(KxN,bf16)) + bias [+ resid]).
// One wave -> 16x64 output strip (4 accumulators), K-step 32.
// LAYOUT: 0 = row-major (M,N); 1 = (B,H,T,HS) scatter; 2 = (B,H,HS,T) scatter.
// oscale folds the attention softmax scale into Q at projection time.
// ---------------------------------------------------------------------------
enum { L_ROW = 0, L_BHTD = 1, L_BHDT = 2 };

template <int LAYOUT, bool RELU, bool RESID, bool OUTBF>
__global__ void __launch_bounds__(128) gemm_bf16_wmma(
    const unsigned short* __restrict__ A,
    const unsigned short* __restrict__ W,
    const float* __restrict__ bias,
    const float* __restrict__ resid,
    void* __restrict__ outp,
    int M, int N, int K, float oscale) {
  const int wave   = threadIdx.x >> 5;
  const int lane   = threadIdx.x & 31;
  const int half   = lane >> 4;
  const int ln     = lane & 15;
  const int m_base = (blockIdx.x * 4 + wave) * 16;
  const int n_base = blockIdx.y * 64;
  if (m_base >= M) return;

  f32x8 acc[4];
#pragma unroll
  for (int i = 0; i < 4; ++i) acc[i] = zero8();

  const unsigned short* arow = A + (size_t)(m_base + ln) * K;
  for (int k0 = 0; k0 < K; k0 += 32) {
    Frag a;
#pragma unroll
    for (int r = 0; r < 8; ++r)
      a.u[r] = *(const unsigned int*)(arow + k0 + kidx(r, half));

    const unsigned short* wrow = W + (size_t)k0 * N + n_base + ln;
#pragma unroll
    for (int nt = 0; nt < 4; ++nt) {
      Frag b;
#pragma unroll
      for (int s = 0; s < 16; ++s)
        b.s[s] = wrow[(size_t)(s + 8 * half + (s & 8)) * N + nt * 16];
      acc[nt] = wmma_bf16(a, b, acc[nt]);
    }
  }

#pragma unroll
  for (int nt = 0; nt < 4; ++nt) {
    const int n  = n_base + nt * 16 + ln;
    const float bv = bias ? bias[n] : 0.0f;
#pragma unroll
    for (int r = 0; r < 8; ++r) {
      const int m = m_base + r + 8 * half;
      float val = acc[nt][r] * oscale + bv;
      if (RESID) val += resid[(size_t)m * N + n];
      if (RELU)  val = fmaxf(val, 0.0f);
      size_t oi;
      if (LAYOUT == L_ROW) {
        oi = (size_t)m * N + n;
      } else {
        const int b = m / TT, tp = m % TT;
        const int h = n >> 6, d = n & 63;
        if (LAYOUT == L_BHTD) oi = (((size_t)b * HH + h) * TT + tp) * HSZ + d;
        else                  oi = (((size_t)b * HH + h) * HSZ + d) * TT + tp;
      }
      if (OUTBF) ((unsigned short*)outp)[oi] = f2bf(val);
      else       ((float*)outp)[oi] = val;
    }
  }
}

// ---------------------------------------------------------------------------
// Flash-style causal attention. One wave per (b, h, 16-row q tile).
// q:(B,H,T,HS) bf16 (pre-scaled by HS^-0.5), kT:(B,H,HS,T) bf16,
// v:(B,H,T,HS) bf16 -> attn:(B,T,C) bf16.
// __launch_bounds__(32, 1): single-wave workgroup, full VGPR budget (no spills).
// ---------------------------------------------------------------------------
__global__ void __launch_bounds__(32, 1) attention_kernel(
    const unsigned short* __restrict__ q,
    const unsigned short* __restrict__ kT,
    const unsigned short* __restrict__ v,
    unsigned short* __restrict__ attn) {
  __shared__ unsigned short Pbuf[16 * 32];

  const int lane = threadIdx.x & 31;
  const int half = lane >> 4;
  const int ln   = lane & 15;
  const int m_base = blockIdx.x * 16;
  const int h = blockIdx.y;
  const int b = blockIdx.z;
  const size_t bh = (size_t)b * HH + h;

  const unsigned short* qp = q  + bh * TT * HSZ;
  const unsigned short* kp = kT + bh * HSZ * TT;
  const unsigned short* vp = v  + bh * TT * HSZ;

  // Q fragments for both 32-wide chunks of HS=64.
  Frag aq[2];
  {
    const unsigned short* qr = qp + (size_t)(m_base + ln) * HSZ;
#pragma unroll
    for (int kc = 0; kc < 2; ++kc)
#pragma unroll
      for (int r = 0; r < 8; ++r)
        aq[kc].u[r] = *(const unsigned int*)(qr + kc * 32 + kidx(r, half));
  }

  f32x8 O[4];
#pragma unroll
  for (int i = 0; i < 4; ++i) O[i] = zero8();
  float mi[8], li[8];
#pragma unroll
  for (int r = 0; r < 8; ++r) { mi[r] = -3.0e38f; li[r] = 0.0f; }

  for (int s0 = 0; s0 <= m_base + 15; s0 += 32) {
    // S = q . k^T  (two 16x16 tiles covering 32 key positions)
    f32x8 S[2];
#pragma unroll
    for (int t = 0; t < 2; ++t) {
      S[t] = zero8();
#pragma unroll
      for (int kc = 0; kc < 2; ++kc) {
        Frag bf;
        const unsigned short* kr = kp + (size_t)(kc * 32) * TT + s0 + t * 16 + ln;
#pragma unroll
        for (int s = 0; s < 16; ++s)
          bf.s[s] = kr[(size_t)(s + 8 * half + (s & 8)) * TT];
        S[t] = wmma_bf16(aq[kc], bf, S[t]);
      }
    }

    // mask + online softmax (rows live in 16-lane groups; scale pre-folded in q)
#pragma unroll
    for (int r = 0; r < 8; ++r) {
      const int m_g = m_base + r + 8 * half;
      float s0v = (s0 + ln      > m_g) ? -1.0e30f : S[0][r];
      float s1v = (s0 + 16 + ln > m_g) ? -1.0e30f : S[1][r];

      float rmax = fmaxf(s0v, s1v);
#pragma unroll
      for (int off = 1; off < 16; off <<= 1)
        rmax = fmaxf(rmax, __shfl_xor(rmax, off, 32));
      const float nm = fmaxf(mi[r], rmax);

      const float p0 = __expf(s0v - nm);
      const float p1 = __expf(s1v - nm);
      float rs = p0 + p1;
#pragma unroll
      for (int off = 1; off < 16; off <<= 1)
        rs += __shfl_xor(rs, off, 32);

      const float corr = __expf(mi[r] - nm);
      li[r] = li[r] * corr + rs;
      mi[r] = nm;
#pragma unroll
      for (int dt = 0; dt < 4; ++dt) O[dt][r] *= corr;

      const int mloc = r + 8 * half;
      Pbuf[mloc * 32 + ln]      = f2bf(p0);
      Pbuf[mloc * 32 + 16 + ln] = f2bf(p1);
    }

    // P (16x32) back as an A-fragment via LDS transpose (same-wave ordering).
    Frag pa;
#pragma unroll
    for (int r = 0; r < 8; ++r)
      pa.u[r] = *(const unsigned int*)(&Pbuf[ln * 32 + kidx(r, half)]);

    // O += P . V
#pragma unroll
    for (int dt = 0; dt < 4; ++dt) {
      Frag bv;
      const unsigned short* vr = vp + (size_t)s0 * HSZ + dt * 16 + ln;
#pragma unroll
      for (int s = 0; s < 16; ++s)
        bv.s[s] = vr[(size_t)(s + 8 * half + (s & 8)) * HSZ];
      O[dt] = wmma_bf16(pa, bv, O[dt]);
    }
  }

  // normalize + write concat-head output (B,T,C)
#pragma unroll
  for (int r = 0; r < 8; ++r) {
    const float inv = 1.0f / li[r];
    const int t_g = m_base + r + 8 * half;
    unsigned short* orow = attn + ((size_t)b * TT + t_g) * CC + h * HSZ;
#pragma unroll
    for (int dt = 0; dt < 4; ++dt)
      orow[dt * 16 + ln] = f2bf(O[dt][r] * inv);
  }
}

// ---------------------------------------------------------------------------
// Host orchestration
// ---------------------------------------------------------------------------
extern "C" void kernel_launch(void* const* d_in, const int* in_sizes, int n_in,
                              void* d_out, int out_size, void* d_ws, size_t ws_size,
                              hipStream_t stream) {
  (void)in_sizes; (void)n_in; (void)out_size; (void)ws_size;

  const float* x   = (const float*)d_in[0];
  const float* Wq  = (const float*)d_in[1];
  const float* Wk  = (const float*)d_in[2];
  const float* Wv  = (const float*)d_in[3];
  const float* Wo  = (const float*)d_in[4];
  const float* bo  = (const float*)d_in[5];
  const float* W1  = (const float*)d_in[6];
  const float* b1  = (const float*)d_in[7];
  const float* W2  = (const float*)d_in[8];
  const float* b2  = (const float*)d_in[9];
  const float* g1  = (const float*)d_in[10];
  const float* be1 = (const float*)d_in[11];
  const float* g2  = (const float*)d_in[12];
  const float* be2 = (const float*)d_in[13];

  char* ws = (char*)d_ws;
  size_t off = 0;
  auto carve = [&](size_t bytes) -> char* {
    char* p = ws + off;
    off += (bytes + 255) & ~(size_t)255;
    return p;
  };

  const size_t actB = (size_t)MM * CC * sizeof(unsigned short);   // 25.2 MB
  unsigned short* wq_bf = (unsigned short*)carve((size_t)CC * CC * 2);
  unsigned short* wk_bf = (unsigned short*)carve((size_t)CC * CC * 2);
  unsigned short* wv_bf = (unsigned short*)carve((size_t)CC * CC * 2);
  unsigned short* wo_bf = (unsigned short*)carve((size_t)CC * CC * 2);
  unsigned short* w1_bf = (unsigned short*)carve((size_t)CC * DFFN * 2);
  unsigned short* w2_bf = (unsigned short*)carve((size_t)DFFN * CC * 2);
  unsigned short* h_bf  = (unsigned short*)carve(actB);           // LN1 out; reused as attn out
  unsigned short* q_bf  = (unsigned short*)carve(actB);           // reused as LN2 out
  unsigned short* kT_bf = (unsigned short*)carve(actB + 8192);    // pad for tail over-read
  unsigned short* v_bf  = (unsigned short*)carve(actB + 8192);    // pad for tail over-read
  float*          x2    = (float*)carve((size_t)MM * CC * 4);     // x + attn@Wo + bo
  unsigned short* ff_bf = (unsigned short*)carve((size_t)MM * DFFN * 2);
  unsigned short* attn_bf = h_bf;   // h_bf is dead once QKV GEMMs complete
  unsigned short* h2_bf   = q_bf;   // q_bf is dead once attention completes

  // --- weight repack to bf16 row-major (K,N) ---
  {
    const int nqkv = CC * CC;
    convert_weight<<<(nqkv + 255) / 256, 256, 0, stream>>>(Wq, wq_bf, CC, CC, HSZ, CC * HSZ, 1);
    convert_weight<<<(nqkv + 255) / 256, 256, 0, stream>>>(Wk, wk_bf, CC, CC, HSZ, CC * HSZ, 1);
    convert_weight<<<(nqkv + 255) / 256, 256, 0, stream>>>(Wv, wv_bf, CC, CC, HSZ, CC * HSZ, 1);
    convert_weight<<<(nqkv + 255) / 256, 256, 0, stream>>>(Wo, wo_bf, CC, CC, CC, 64, 1);
    const int n1 = CC * DFFN;
    convert_weight<<<(n1 + 255) / 256, 256, 0, stream>>>(W1, w1_bf, CC, DFFN, DFFN, 64, 1);
    convert_weight<<<(n1 + 255) / 256, 256, 0, stream>>>(W2, w2_bf, DFFN, CC, CC, 64, 1);
  }

  // --- LN1 ---
  layernorm_bf16<<<MM, 128, 0, stream>>>(x, g1, be1, h_bf);

  // --- QKV projections (q pre-scaled by HS^-0.5; k stored transposed) ---
  const dim3 gC(MM / 64, CC / 64);
  gemm_bf16_wmma<L_BHTD, false, false, true><<<gC, 128, 0, stream>>>(
      h_bf, wq_bf, nullptr, nullptr, q_bf, MM, CC, CC, 0.125f);
  gemm_bf16_wmma<L_BHDT, false, false, true><<<gC, 128, 0, stream>>>(
      h_bf, wk_bf, nullptr, nullptr, kT_bf, MM, CC, CC, 1.0f);
  gemm_bf16_wmma<L_BHTD, false, false, true><<<gC, 128, 0, stream>>>(
      h_bf, wv_bf, nullptr, nullptr, v_bf, MM, CC, CC, 1.0f);

  // --- flash-style causal attention ---
  attention_kernel<<<dim3(TT / 16, HH, BB), 32, 0, stream>>>(q_bf, kT_bf, v_bf, attn_bf);

  // --- output projection + residual (f32) ---
  gemm_bf16_wmma<L_ROW, false, true, false><<<gC, 128, 0, stream>>>(
      attn_bf, wo_bf, bo, x, x2, MM, CC, CC, 1.0f);

  // --- LN2 ---
  layernorm_bf16<<<MM, 128, 0, stream>>>(x2, g2, be2, h2_bf);

  // --- FFN1 (ReLU) ---
  gemm_bf16_wmma<L_ROW, true, false, true><<<dim3(MM / 64, DFFN / 64), 128, 0, stream>>>(
      h2_bf, w1_bf, b1, nullptr, ff_bf, MM, DFFN, CC, 1.0f);

  // --- FFN2 + residual -> d_out (f32) ---
  gemm_bf16_wmma<L_ROW, false, true, false><<<gC, 128, 0, stream>>>(
      ff_bf, w2_bf, b2, x2, (float*)d_out, MM, CC, DFFN, 1.0f);
}
